// AttentionRNN_82051055222920
// MI455X (gfx1250) — matrix operand order repaired and checked by reading, not compiled, not run on hardware
//
#include <hip/hip_runtime.h>
#include <hip/hip_bf16.h>
#include <stdint.h>

typedef __attribute__((ext_vector_type(16))) _Float16 v16h;
typedef __attribute__((ext_vector_type(8)))  float    v8f;

#define B_    64
#define NUM_  8
#define TC_   64
#define H_    128
#define BN_   (B_ * NUM_)     // 512
#define H3_   (3 * H_)        // 384
#define LOG2E 1.4426950408889634f

// ---------- fast transcendentals on the hardware trans pipe ----------
__device__ __forceinline__ float fast_exp(float x) {
  return __builtin_amdgcn_exp2f(x * LOG2E);
}
__device__ __forceinline__ float fast_tanh(float x) {
  float t = __builtin_amdgcn_exp2f(x * (2.0f * LOG2E));
  return (t - 1.0f) * __builtin_amdgcn_rcpf(t + 1.0f);
}
__device__ __forceinline__ float fast_sigmoid(float x) {
  return __builtin_amdgcn_rcpf(1.0f + __builtin_amdgcn_exp2f(-x * LOG2E));
}

// ---------- WMMA fragment layout helpers ----------
// 16-bit A-matrix 16x32, contiguous k order chosen by this code: lane (l,hi) element e holds
// K = e + hi*16.  Inverse: given kk in [0,32):
//   hi = kk>>4 ;  e = kk & 15
// Fragment-ordered storage: block[kt] is 32 lanes x 16 halfs contiguous.
__device__ __forceinline__ int a_off(int row_l, int kk) {
  int hi = kk >> 4;
  int e  = kk & 15;
  return (row_l + hi * 16) * 16 + e;
}
// 16-bit B-matrix 32x16, same contiguous k order as A: lane (l,hi) element e holds K = e + hi*16, N = l.
__device__ __forceinline__ int b_off(int l, int kk) {
  return (l + (kk >> 4) * 16) * 16 + (kk & 15);
}
// Fragment-ordered packed B (global): one contiguous v16h per lane.
__device__ __forceinline__ v16h load_b_packed(const _Float16* base, int kt, int nt, int NT, int lane) {
  const v16h* p = (const v16h*)base;
  return p[(kt * NT + nt) * 32 + lane];
}

// ---------- async global->LDS staging (CDNA5) ----------
__device__ __forceinline__ void async_load_b128(void* lds_dst, const void* gaddr) {
  unsigned lds = (unsigned)(uintptr_t)lds_dst;  // low 32 bits of flat LDS ptr = DS offset
  asm volatile("global_load_async_to_lds_b128 %0, %1, off"
               :: "v"(lds), "v"(gaddr) : "memory");
}
__device__ __forceinline__ void wait_asynccnt0() {
  asm volatile("s_wait_asynccnt 0" ::: "memory");
}

// ---------- kernel 0: pack f32 KxN weight into fragment-ordered f16 B ----------
__global__ __launch_bounds__(32) void pack_b_kernel(const float* __restrict__ src,
                                                    _Float16* __restrict__ dst,
                                                    int N, int NT) {
  int kt   = blockIdx.x / NT;
  int nt   = blockIdx.x % NT;
  int lane = threadIdx.x;
  int l  = lane & 15;
  int hi = lane >> 4;
  _Float16* d = dst + ((size_t)blockIdx.x * 32 + lane) * 16;
#pragma unroll
  for (int e = 0; e < 16; e++) {
    int k = kt * 32 + e + hi * 16;
    d[e] = (_Float16)src[k * N + nt * 16 + l];
  }
}

// ---------- kernel 1: uproj[b', i, s] = sum_t x[b', t, i] * att_u[t, s] ----------
__global__ __launch_bounds__(256) void uproj_kernel(const float* __restrict__ x,
                                                    const float* __restrict__ att_u,
                                                    float* __restrict__ up) {
  // fragment-ordered LDS: AsA[band(8)][kt(2)][lane(32)][e(16)], UsB[kt(2)][nt(4)][lane][e]
  __shared__ __align__(32) _Float16 AsA[8 * 2 * 512];
  __shared__ __align__(32) _Float16 UsB[2 * 4 * 512];
  const int tid = threadIdx.x;
  const int bp  = blockIdx.x;
  const float* xb = x + (size_t)bp * TC_ * H_;
#pragma unroll
  for (int q = 0; q < 32; q++) {
    int idx = q * 256 + tid;
    int s = idx >> 7, i = idx & 127;          // value = x[b'][s][i]; A row = i, K = s
    int band = i >> 4, row_l = i & 15;
    int kt = s >> 5, kk = s & 31;
    AsA[(band * 2 + kt) * 512 + a_off(row_l, kk)] = (_Float16)xb[idx];
  }
#pragma unroll
  for (int q = 0; q < 16; q++) {
    int idx = q * 256 + tid;
    int tK = idx >> 6, sN = idx & 63;         // att_u[tK][sN]; B row K = tK, N = sN
    int kt = tK >> 5, kk = tK & 31;
    int nt = sN >> 4, l = sN & 15;
    UsB[(kt * 4 + nt) * 512 + b_off(l, kk)] = (_Float16)att_u[idx];
  }
  __syncthreads();
  const int wv = tid >> 5, lane = tid & 31, l = lane & 15, hi = lane >> 4;
  const int i0 = wv * 16;
  const v16h* Aa = (const v16h*)AsA;
  const v16h* Bb = (const v16h*)UsB;
  float* upb = up + (size_t)bp * H_ * TC_;
  for (int nt = 0; nt < 4; nt++) {
    v8f acc = {};
#pragma unroll
    for (int kt = 0; kt < 2; kt++) {
      v16h a = Aa[(wv * 2 + kt) * 32 + lane];
      v16h b = Bb[(kt * 4 + nt) * 32 + lane];
      acc = __builtin_amdgcn_wmma_f32_16x16x32_f16(false, a, false, b, (short)0, acc, false, false);
    }
#pragma unroll
    for (int r = 0; r < 8; r++) upb[(i0 + r + hi * 8) * TC_ + nt * 16 + l] = acc[r];
  }
}

// ---------- kernel 2: persistent GRU scan, 16 rows per workgroup ----------
__global__ __launch_bounds__(256) void gru_persist_kernel(
    const float* __restrict__ x, const float* __restrict__ att_v,
    const float* __restrict__ gbias, const float* __restrict__ up,
    const _Float16* __restrict__ WkP, const _Float16* __restrict__ RkP,
    const _Float16* __restrict__ AwP, float* __restrict__ out) {
  __shared__ float    hs32[16 * H_];
  __shared__ __align__(32) _Float16 hsA[4 * 512];   // h_state, A-fragment order (K=128 -> 4 blocks)
  __shared__ __align__(32) _Float16 xtA[4 * 512];   // x_t,     A-fragment order
  __shared__ float    xcol32[16 * H_];              // async-staged x column (f32)
  __shared__ float    hp[16 * TC_];
  __shared__ float    ea[16 * H_];
  __shared__ float    xm[16 * H3_];
  __shared__ float    rm[16 * H3_];
  __shared__ float    redm[16 * 8];
  __shared__ float    reds[16 * 8];
  __shared__ float    rmax_[16];
  __shared__ float    rsum_[16];
  __shared__ float    vv[TC_];

  const int tid  = threadIdx.x;
  const int wv   = tid >> 5;
  const int lane = tid & 31;
  const int l    = lane & 15;
  const int hi   = lane >> 4;
  const int bp0  = blockIdx.x * 16;

  if (tid < TC_) vv[tid] = att_v[tid];
#pragma unroll
  for (int q = 0; q < 8; q++) {
    int idx = q * 256 + tid;
    hs32[idx] = 0.0f;
    hsA[idx & 2047] = (_Float16)0.0f;  // 2048 halfs zeroed (idx 0..2047 covered)
  }
  __syncthreads();

  // step-invariant register cache of v
  float vvc[TC_];
#pragma unroll
  for (int s = 0; s < TC_; s++) vvc[s] = vv[s];

  const v16h* HsA = (const v16h*)hsA;
  const v16h* XtA = (const v16h*)xtA;
  const int br_e = tid >> 4;   // phase-1 row (0..15), 16 threads per row
  const int ie0  = tid & 15;

  for (int t = 0; t < TC_; t++) {
    // ---- phase 0: waves 0-3: hp = h @ att_w ; waves 4-7: async-stage x_col
    if (wv < 4) {
      v8f acc = {};
#pragma unroll
      for (int kt = 0; kt < 4; kt++) {
        v16h a = HsA[kt * 32 + lane];
        v16h b = load_b_packed(AwP, kt, wv, 4, lane);
        acc = __builtin_amdgcn_wmma_f32_16x16x32_f16(false, a, false, b, (short)0, acc, false, false);
      }
#pragma unroll
      for (int r = 0; r < 8; r++) hp[(r + hi * 8) * TC_ + wv * 16 + l] = acc[r];
    } else {
      int tt = tid - 128;
#pragma unroll
      for (int q = 0; q < 4; q++) {
        int c  = q * 128 + tt;            // 512 chunks of 16B
        int br = c >> 5;
        int i4 = (c & 31) * 4;
        async_load_b128(&xcol32[br * H_ + i4],
                        x + ((size_t)(bp0 + br) * TC_ + t) * H_ + i4);
      }
    }
    __syncthreads();   // hp ready (async x_col still in flight)

    // ---- phase 1: e[b,i] = sum_s tanh(hp[b,s] + uproj[b,i,s]) * v[s]
    {
      float hpc[TC_];
#pragma unroll
      for (int s4 = 0; s4 < 16; s4++) {
        float4 h4 = *(const float4*)(hp + br_e * TC_ + s4 * 4);
        hpc[s4 * 4 + 0] = h4.x; hpc[s4 * 4 + 1] = h4.y;
        hpc[s4 * 4 + 2] = h4.z; hpc[s4 * 4 + 3] = h4.w;
      }
      for (int w = 0; w < 8; w++) {
        int i = ie0 + 16 * w;
        const float4* u4 = (const float4*)(up + ((size_t)(bp0 + br_e) * H_ + i) * TC_);
        float acc = 0.0f;
#pragma unroll
        for (int s4 = 0; s4 < 16; s4++) {
          float4 uu = u4[s4];
          acc += fast_tanh(hpc[s4 * 4 + 0] + uu.x) * vvc[s4 * 4 + 0];
          acc += fast_tanh(hpc[s4 * 4 + 1] + uu.y) * vvc[s4 * 4 + 1];
          acc += fast_tanh(hpc[s4 * 4 + 2] + uu.z) * vvc[s4 * 4 + 2];
          acc += fast_tanh(hpc[s4 * 4 + 3] + uu.w) * vvc[s4 * 4 + 3];
        }
        ea[br_e * H_ + i] = acc;
      }
    }
    __syncthreads();

    // ---- phase 2: softmax over i (128) per row
    if (tid < 128) {
      int br = tid >> 3, part = tid & 7;
      const float* er = ea + br * H_ + part * 16;
      float m = er[0];
#pragma unroll
      for (int j = 1; j < 16; j++) m = fmaxf(m, er[j]);
      redm[br * 8 + part] = m;
    }
    __syncthreads();
    if (tid < 16) {
      float m = redm[tid * 8];
#pragma unroll
      for (int j = 1; j < 8; j++) m = fmaxf(m, redm[tid * 8 + j]);
      rmax_[tid] = m;
    }
    __syncthreads();
    if (tid < 128) {
      int br = tid >> 3, part = tid & 7;
      float m = rmax_[br];
      float* er = ea + br * H_ + part * 16;
      float s = 0.0f;
#pragma unroll
      for (int j = 0; j < 16; j++) {
        float ev = fast_exp(er[j] - m);
        er[j] = ev;
        s += ev;
      }
      reds[br * 8 + part] = s;
    }
    __syncthreads();
    if (tid < 16) {
      float s = 0.0f;
#pragma unroll
      for (int j = 0; j < 8; j++) s += reds[tid * 8 + j];
      rsum_[tid] = __builtin_amdgcn_rcpf(s);
    }
    // issuing waves drain their async copies before the barrier publishing xcol32
    if (wv >= 4) wait_asynccnt0();
    __syncthreads();

    // ---- phase 3: x_t = x_col * a, scattered into A-fragment order
#pragma unroll
    for (int q = 0; q < 8; q++) {
      int idx = q * 256 + tid;
      int br = idx >> 7, i = idx & 127;
      float a_ = ea[idx] * rsum_[br];
      xtA[(i >> 5) * 512 + a_off(br, i & 31)] = (_Float16)(xcol32[idx] * a_);
    }
    __syncthreads();

    // ---- phase 4: xm = x_t @ gru_kernel + b_in ; rm = h @ gru_rkernel + b_rec
    for (int T = wv; T < 48; T += 8) {
      bool isx = (T < 24);
      int nt = isx ? T : (T - 24);
      const v16h* Aa = isx ? XtA : HsA;
      const _Float16* Bp = isx ? WkP : RkP;
      v8f acc = {};
#pragma unroll
      for (int kt = 0; kt < 4; kt++) {
        v16h a = Aa[kt * 32 + lane];
        v16h b = load_b_packed(Bp, kt, nt, 24, lane);
        acc = __builtin_amdgcn_wmma_f32_16x16x32_f16(false, a, false, b, (short)0, acc, false, false);
      }
      float bias = gbias[(isx ? 0 : H3_) + nt * 16 + l];
      float* Ob = isx ? xm : rm;
#pragma unroll
      for (int r = 0; r < 8; r++) Ob[(r + hi * 8) * H3_ + nt * 16 + l] = acc[r] + bias;
    }
    __syncthreads();

    // ---- phase 5: gates + state update (writes h back in fragment order)
#pragma unroll
    for (int q = 0; q < 8; q++) {
      int idx = q * 256 + tid;
      int br = idx >> 7, j = idx & 127;
      const float* xr = xm + br * H3_;
      const float* rr = rm + br * H3_;
      float z  = fast_sigmoid(xr[j] + rr[j]);
      float r_ = fast_sigmoid(xr[128 + j] + rr[128 + j]);
      float hh = xr[256 + j] + r_ * rr[256 + j];
      hh = fmaxf(hh, 0.0f);
      float hnew = z * hs32[idx] + (1.0f - z) * hh;
      hs32[idx] = hnew;
      hsA[(j >> 5) * 512 + a_off(br, j & 31)] = (_Float16)hnew;
    }
    __syncthreads();
  }

  // ---- write final h
#pragma unroll
  for (int q = 0; q < 8; q++) {
    int idx = q * 256 + tid;
    out[(size_t)bp0 * H_ + idx] = hs32[idx];
  }
}

extern "C" void kernel_launch(void* const* d_in, const int* in_sizes, int n_in,
                              void* d_out, int out_size, void* d_ws, size_t ws_size,
                              hipStream_t stream) {
  (void)in_sizes; (void)n_in; (void)out_size; (void)ws_size;
  const float* x    = (const float*)d_in[0];
  const float* attv = (const float*)d_in[1];  // (TC,1)
  const float* attw = (const float*)d_in[2];  // (H,TC)
  const float* attu = (const float*)d_in[3];  // (TC,TC)
  const float* gk   = (const float*)d_in[4];  // (H,3H)
  const float* grk  = (const float*)d_in[5];  // (H,3H)
  const float* gb   = (const float*)d_in[6];  // (2,3H)
  float* out = (float*)d_out;

  // workspace: uproj (16 MiB f32) | WkP | RkP | AwP (fragment-packed f16)
  char* ws = (char*)d_ws;
  float* up = (float*)ws;
  size_t up_bytes = (size_t)BN_ * H_ * TC_ * sizeof(float);
  _Float16* WkP = (_Float16*)(ws + up_bytes);
  _Float16* RkP = WkP + (size_t)4 * 24 * 32 * 16;
  _Float16* AwP = RkP + (size_t)4 * 24 * 32 * 16;

  pack_b_kernel<<<4 * 24, 32, 0, stream>>>(gk,   WkP, H3_, 24);
  pack_b_kernel<<<4 * 24, 32, 0, stream>>>(grk,  RkP, H3_, 24);
  pack_b_kernel<<<4 * 4,  32, 0, stream>>>(attw, AwP, TC_, 4);
  uproj_kernel<<<BN_, 256, 0, stream>>>(x, attu, up);
  gru_persist_kernel<<<BN_ / 16, 256, 0, stream>>>(x, attv, gb, up, WkP, RkP, AwP, out);
}
